// CompatibleEnhancedUFormer_69818988363879
// MI455X (gfx1250) — matrix operand order repaired and checked by reading, not compiled, not run on hardware
//
#include <hip/hip_runtime.h>
#include <cstdint>

// ---------------------------------------------------------------------------
// CDNA5 (gfx1250) UFormer forward pass — v2.
//  * all GEMMs on v_wmma_f32_16x16x32_f16 (f16 A/B, f32 accum)
//  * weights converted per-launch into WMMA-fragment-swizzled f16 so B
//    fragments are two coalesced global_load_b128 (no LDS for B)
//  * A staged to LDS with float4 loads + packed f16 b64 stores (no guards;
//    M % 64 == 0 and K % 32 == 0 by construction)
// ---------------------------------------------------------------------------

typedef _Float16 h16;
typedef __attribute__((ext_vector_type(4)))  _Float16 v4h;
typedef __attribute__((ext_vector_type(8)))  _Float16 v8h;
typedef __attribute__((ext_vector_type(16))) _Float16 v16h;
typedef __attribute__((ext_vector_type(8)))  float    v8f;

#define WMMA_F16(a, b, c) \
  __builtin_amdgcn_wmma_f32_16x16x32_f16(false, (a), false, (b), (short)0, (c), false, false)

// ---- WMMA fragment helpers (wave32; A/B k = 16*(lane/16) + element) -----------

// A fragment 16x32 from row-major f16 LDS tile (ld halves). Two b128 loads.
__device__ __forceinline__ v16h frag_a(const h16* s, int ld) {
  int lane = threadIdx.x & 31;
  int row  = lane & 15;
  int kb   = (lane >> 4) ? 16 : 0;
  const h16* p = s + row * ld + kb;
  v8h lo = *(const v8h*)(p);
  v8h hi = *(const v8h*)(p + 8);
  return __builtin_shufflevector(lo, hi, 0,1,2,3,4,5,6,7,8,9,10,11,12,13,14,15);
}

// B fragment from a row-major [n][k] tile (tile^T as B); contiguous 16 halves.
__device__ __forceinline__ v16h frag_bt(const h16* s, int ld) {
  int lane = threadIdx.x & 31;
  int col  = lane & 15;
  int kb   = (lane >> 4) ? 16 : 0;
  const h16* p = s + col * ld + kb;
  v8h lo = *(const v8h*)(p);
  v8h hi = *(const v8h*)(p + 8);
  return __builtin_shufflevector(lo, hi, 0,1,2,3,4,5,6,7,8,9,10,11,12,13,14,15);
}

// B fragment from pre-swizzled global weights: block of 512 halves per
// (ktile, ntile); lane's 16 halves are contiguous -> two global_load_b128.
__device__ __forceinline__ v16h frag_bg(const h16* p) {
  int lane = threadIdx.x & 31;
  const h16* q = p + lane * 16;
  v8h lo = *(const v8h*)(q);
  v8h hi = *(const v8h*)(q + 8);
  return __builtin_shufflevector(lo, hi, 0,1,2,3,4,5,6,7,8,9,10,11,12,13,14,15);
}

// D/C 16x16 f32 store to LDS: vgpr r -> M=r (+8 for lanes 16-31), N=lane&15.
__device__ __forceinline__ void store_c_f32(float* d, int ld, v8f c) {
  int lane = threadIdx.x & 31;
  int n = lane & 15, mh = (lane >> 4) * 8;
#pragma unroll
  for (int r = 0; r < 8; r++) d[(mh + r) * ld + n] = c[r];
}

// ---- weight conversion: f32 [N,K] -> f16 swizzled fragment order -----------
// Wt layout: ((kt*nT16 + ntile)*32 + lane)*16 + i ; k = kt*32 + 16*(lane>>4) + i, n = ntile*16 + (lane&15).
__global__ __launch_bounds__(256) void k_wt(const float* __restrict__ W,
                                            h16* __restrict__ Wt,
                                            int N, int Ksrc, int Npad, int total) {
  int idx = blockIdx.x * 256 + threadIdx.x;
  if (idx >= total) return;
  int i = idx & 15; int lane = (idx >> 4) & 31; int t = idx >> 9;
  int nT16 = Npad >> 4;
  int ntile = t % nT16; int kt = t / nT16;
  int col = lane & 15; int kb = (lane >> 4) ? 16 : 0;
  int k = kt * 32 + kb + i; int nn = ntile * 16 + col;
  h16 v = (h16)0.f;
  if (k < Ksrc && nn < N) v = (h16)W[(size_t)nn * Ksrc + k];
  Wt[idx] = v;
}

// ConvTranspose weight [Cin, O, KK] -> swizzled f16 with col n = pq*O + o.
__global__ __launch_bounds__(256) void k_wt_up(const float* __restrict__ W,
                                               h16* __restrict__ Wt,
                                               int Cin, int O, int KK,
                                               int Npad, int total) {
  int idx = blockIdx.x * 256 + threadIdx.x;
  if (idx >= total) return;
  int i = idx & 15; int lane = (idx >> 4) & 31; int t = idx >> 9;
  int nT16 = Npad >> 4;
  int ntile = t % nT16; int kt = t / nT16;
  int col = lane & 15; int kb = (lane >> 4) ? 16 : 0;
  int k = kt * 32 + kb + i; int nn = ntile * 16 + col;
  h16 v = (h16)0.f;
  int N = KK * O;
  if (k < Cin && nn < N) {
    int pq = nn / O; int o = nn % O;
    v = (h16)W[((size_t)k * O + o) * KK + pq];
  }
  Wt[idx] = v;
}

// ---- main GEMM: C[M,N] = act(A_f32[M,K] x W_f16[K,N] + bias (+resid)) ------
// Requires M % 64 == 0, K % 32 == 0. 256 threads -> 64x64 C tile, 8 waves x 2
// subtiles. B fragments straight from swizzled global weights.
__global__ __launch_bounds__(256) void k_gemm(const float* __restrict__ A,
                                              const h16* __restrict__ Wt,
                                              const float* __restrict__ bias,
                                              const float* __restrict__ resid,
                                              float* __restrict__ C,
                                              int M, int N, int K, int Npad,
                                              int act, int biasPeriod) {
  __shared__ h16 As[64 * 40];   // 64 x 32, ld 40
  const int tid  = threadIdx.x;
  const int m0   = blockIdx.y * 64, n0 = blockIdx.x * 64;
  const int wave = tid >> 5;
  const int s0 = wave, s1 = wave + 8;
  const int mt0 = (s0 >> 2) * 16, nt0 = (s0 & 3) * 16;
  const int mt1 = (s1 >> 2) * 16, nt1 = (s1 & 3) * 16;
  const int nT16 = Npad >> 4;
  const int ntg0 = (n0 + nt0) >> 4, ntg1 = (n0 + nt1) >> 4;
  v8f acc0 = {}, acc1 = {};

  for (int k0 = 0; k0 < K; k0 += 32) {
    const int kt = k0 >> 5;
#pragma unroll
    for (int i = 0; i < 2; i++) {
      int e = (tid + 256 * i) * 4;           // 2048 floats, 4 at a time
      int r = e >> 5, k = e & 31;
      const float4 f = *(const float4*)(A + (size_t)(m0 + r) * K + k0 + k);
      v4h hv = {(h16)f.x, (h16)f.y, (h16)f.z, (h16)f.w};
      *(v4h*)(&As[r * 40 + k]) = hv;
    }
    __syncthreads();
    v16h b0 = frag_bg(Wt + ((size_t)kt * nT16 + ntg0) * 512);
    v16h b1 = frag_bg(Wt + ((size_t)kt * nT16 + ntg1) * 512);
    v16h a0 = frag_a(As + mt0 * 40, 40);
    acc0 = WMMA_F16(a0, b0, acc0);
    v16h a1 = frag_a(As + mt1 * 40, 40);
    acc1 = WMMA_F16(a1, b1, acc1);
    __syncthreads();
  }

  const int lane = tid & 31;
  const int n = lane & 15, mh = (lane >> 4) * 8;
  {
    int gn = n0 + nt0 + n;
    if (gn < N) {
      float bv = bias[gn % biasPeriod];
#pragma unroll
      for (int r = 0; r < 8; r++) {
        int gm = m0 + mt0 + mh + r;
        float v = acc0[r] + bv;
        if (resid) v += resid[(size_t)gm * N + gn];
        if (act == 1) v = 0.5f * v * (1.0f + erff(v * 0.70710678118f));
        C[(size_t)gm * N + gn] = v;
      }
    }
  }
  {
    int gn = n0 + nt1 + n;
    if (gn < N) {
      float bv = bias[gn % biasPeriod];
#pragma unroll
      for (int r = 0; r < 8; r++) {
        int gm = m0 + mt1 + mh + r;
        float v = acc1[r] + bv;
        if (resid) v += resid[(size_t)gm * N + gn];
        if (act == 1) v = 0.5f * v * (1.0f + erff(v * 0.70710678118f));
        C[(size_t)gm * N + gn] = v;
      }
    }
  }
}

// ---- fused windowed attention: one wave per (window, head), d = 32 ---------
__global__ __launch_bounds__(32) void k_attn(const float* __restrict__ qkv,
                                             float* __restrict__ out,
                                             int B, int H, int W, int C,
                                             int heads, int shift) {
  __shared__ h16   Qs[64 * 40];   // [token][e]   (A operand, row-major)
  __shared__ h16   Ks[64 * 40];   // [token][e]   (B via frag_bt: K^T)
  __shared__ h16   Vt[32 * 72];   // [e][token]   (B via frag_bt: contiguous k)
  __shared__ float Ss[64 * 68];   // scores f32
  __shared__ h16   Am[64 * 72];   // softmax(A) f16, row-major
  int lane = threadIdx.x;
  int wid  = blockIdx.x;
  int nh = H >> 3, nw = W >> 3;
  int head = wid % heads; int t = wid / heads;
  int ww = t % nw; t /= nw; int wh = t % nh; int bb = t / nh;
  const float scale = 0.17677669529663687f;  // 1/sqrt(32)
  int C3 = 3 * C;

  // gather (shift handled by modular indexing == jnp.roll), float4 loads
#pragma unroll
  for (int rep = 0; rep < 2; rep++) {
    int tk = lane + rep * 32;
    int r = tk >> 3, c = tk & 7;
    int h = (wh * 8 + r + shift) % H;
    int w = (ww * 8 + c + shift) % W;
    const float* p = qkv + (size_t)((bb * H + h) * W + w) * C3;
    const float4* pq = (const float4*)(p + head * 32);
    const float4* pk = (const float4*)(p + (heads + head) * 32);
    const float4* pv = (const float4*)(p + (2 * heads + head) * 32);
#pragma unroll
    for (int u = 0; u < 8; u++) {
      float4 fq = pq[u], fk = pk[u], fv = pv[u];
      v4h hq = {(h16)(fq.x * scale), (h16)(fq.y * scale),
                (h16)(fq.z * scale), (h16)(fq.w * scale)};
      *(v4h*)(&Qs[tk * 40 + u * 4]) = hq;
      v4h hk = {(h16)fk.x, (h16)fk.y, (h16)fk.z, (h16)fk.w};
      *(v4h*)(&Ks[tk * 40 + u * 4]) = hk;
      Vt[(u * 4 + 0) * 72 + tk] = (h16)fv.x;
      Vt[(u * 4 + 1) * 72 + tk] = (h16)fv.y;
      Vt[(u * 4 + 2) * 72 + tk] = (h16)fv.z;
      Vt[(u * 4 + 3) * 72 + tk] = (h16)fv.w;
    }
  }
  __syncthreads();

  // S = (Q*scale) K^T   [64x64], K-dim 32 -> one WMMA per 16x16 tile
#pragma unroll
  for (int mt = 0; mt < 4; mt++) {
    v16h a = frag_a(Qs + mt * 16 * 40, 40);
#pragma unroll
    for (int nt = 0; nt < 4; nt++) {
      v8f acc = {};
      v16h b = frag_bt(Ks + nt * 16 * 40, 40);
      acc = WMMA_F16(a, b, acc);
      store_c_f32(Ss + mt * 16 * 68 + nt * 16, 68, acc);
    }
  }
  __syncthreads();

  // row softmax in f32, write A as f16
#pragma unroll
  for (int rep = 0; rep < 2; rep++) {
    int row = lane + rep * 32;
    float* sr = Ss + row * 68;
    float mx = -3.4e38f;
    for (int j = 0; j < 64; j++) mx = fmaxf(mx, sr[j]);
    float sum = 0.f;
    for (int j = 0; j < 64; j++) { float e = __expf(sr[j] - mx); sr[j] = e; sum += e; }
    float inv = 1.f / sum;
    for (int j = 0; j < 64; j++) Am[row * 72 + j] = (h16)(sr[j] * inv);
  }
  __syncthreads();

  // O = A V   [64x32], K-dim 64 -> 2 WMMA K-steps; V fragments contiguous
#pragma unroll
  for (int mt = 0; mt < 4; mt++) {
#pragma unroll
    for (int nt = 0; nt < 2; nt++) {
      v8f acc = {};
#pragma unroll
      for (int ks = 0; ks < 2; ks++) {
        v16h a = frag_a(Am + mt * 16 * 72 + ks * 32, 72);
        v16h b = frag_bt(Vt + nt * 16 * 72 + ks * 32, 72);
        acc = WMMA_F16(a, b, acc);
      }
      int n = lane & 15, mh = (lane >> 4) * 8;
#pragma unroll
      for (int r8 = 0; r8 < 8; r8++) {
        int tk = mt * 16 + mh + r8;
        int rr = tk >> 3, cc = tk & 7;
        int h = (wh * 8 + rr + shift) % H, w = (ww * 8 + cc + shift) % W;
        out[(size_t)((bb * H + h) * W + w) * C + head * 32 + nt * 16 + n] = acc[r8];
      }
    }
  }
}

// ---- LayerNorm over channel dim (token-major), one wave per token ----------
__global__ __launch_bounds__(256) void k_ln(const float* __restrict__ x,
                                            const float* __restrict__ g,
                                            const float* __restrict__ bta,
                                            float* __restrict__ y, int M, int C) {
  int tok  = blockIdx.x * 8 + (threadIdx.x >> 5);
  int lane = threadIdx.x & 31;
  if (tok >= M) return;
  const float* p = x + (size_t)tok * C;
  float s = 0.f;
  for (int c = lane; c < C; c += 32) s += p[c];
#pragma unroll
  for (int o = 16; o > 0; o >>= 1) s += __shfl_xor(s, o, 32);
  float mu = s / (float)C;
  float v = 0.f;
  for (int c = lane; c < C; c += 32) { float d = p[c] - mu; v += d * d; }
#pragma unroll
  for (int o = 16; o > 0; o >>= 1) v += __shfl_xor(v, o, 32);
  float inv = rsqrtf(v / (float)C + 1e-6f);
  float* yo = y + (size_t)tok * C;
  for (int c = lane; c < C; c += 32) yo[c] = (p[c] - mu) * inv * g[c] + bta[c];
}

// ---- data movement helpers -------------------------------------------------
// NCHW input -> patch im2col [B*64*64, 64] (K padded 48->64 with zeros)
__global__ __launch_bounds__(256) void k_patch(const float* __restrict__ x,
                                               float* __restrict__ y, int B, int total) {
  int idx = blockIdx.x * 256 + threadIdx.x;
  if (idx >= total) return;
  int k = idx & 63; int token = idx >> 6;
  float v = 0.f;
  if (k < 48) {
    int b = token / 4096; int r = token % 4096; int oh = r >> 6, ow = r & 63;
    int c = k >> 4; int pq = k & 15; int p = pq >> 2, q = pq & 3;
    v = x[((size_t)(b * 3 + c) * 256 + (oh * 4 + p)) * 256 + (ow * 4 + q)];
  }
  (void)B;
  y[idx] = v;
}

// BHWC -> space-to-depth im2col [B*(H/2)*(W/2), 4C], k = c*4 + p*2 + q
__global__ __launch_bounds__(256) void k_s2d(const float* __restrict__ xm,
                                             float* __restrict__ y,
                                             int B, int H, int W, int C, int total) {
  int idx = blockIdx.x * 256 + threadIdx.x;
  if (idx >= total) return;
  int K4 = 4 * C;
  int k = idx % K4; int token = idx / K4;
  int oH = H >> 1, oW = W >> 1;
  int b = token / (oH * oW); int r = token % (oH * oW); int oh = r / oW, ow = r % oW;
  int c = k >> 2; int pq = k & 3; int p = pq >> 1, q = pq & 1;
  (void)B;
  y[idx] = xm[((size_t)((b * H + oh * 2 + p) * W) + (ow * 2 + q)) * C + c];
}

// Y [B,H,W,S*S*O] -> out [B,H*S,W*S,O]  (transposed-conv scatter)
__global__ __launch_bounds__(256) void k_upsc(const float* __restrict__ Y,
                                              float* __restrict__ out,
                                              int B, int H, int W, int O, int S, int total) {
  int idx = blockIdx.x * 256 + threadIdx.x;
  if (idx >= total) return;
  int o = idx % O; int t = idx / O;
  int W2 = W * S, H2 = H * S;
  int ow = t % W2; t /= W2; int oh = t % H2; int b = t / H2;
  int h = oh / S, p = oh % S, w = ow / S, q = ow % S;
  out[idx] = Y[((size_t)((b * H + h) * W + w)) * (S * S * O) + (p * S + q) * O + o];
}

__global__ __launch_bounds__(256) void k_cat(const float* __restrict__ a,
                                             const float* __restrict__ bsrc,
                                             float* __restrict__ y,
                                             int M, int C1, int C2, int total) {
  int idx = blockIdx.x * 256 + threadIdx.x;
  if (idx >= total) return;
  int CT = C1 + C2;
  int c = idx % CT; int m = idx / CT;
  (void)M;
  y[idx] = (c < C1) ? a[(size_t)m * C1 + c] : bsrc[(size_t)m * C2 + (c - C1)];
}

// direct 3x3 conv (pad 1) + sigmoid; x BHWC, W [OC,C,3,3], out NCHW
__global__ __launch_bounds__(256) void k_conv3(const float* __restrict__ x,
                                               const float* __restrict__ Wt,
                                               const float* __restrict__ bias,
                                               float* __restrict__ out,
                                               int B, int H, int W, int C, int OC, int total) {
  int idx = blockIdx.x * 256 + threadIdx.x;
  if (idx >= total) return;
  int w = idx % W; int t = idx / W; int h = t % H; t /= H; int oc = t % OC; int b = t / OC;
  (void)B;
  float s = bias[oc];
  for (int kh = 0; kh < 3; kh++) {
    int ih = h + kh - 1; if ((unsigned)ih >= (unsigned)H) continue;
    for (int kw = 0; kw < 3; kw++) {
      int iw = w + kw - 1; if ((unsigned)iw >= (unsigned)W) continue;
      const float* px = x + ((size_t)(b * H + ih) * W + iw) * C;
      const float* pw = Wt + (size_t)oc * C * 9 + kh * 3 + kw;
      for (int c = 0; c < C; c++) s += px[c] * pw[(size_t)c * 9];
    }
  }
  out[idx] = 1.f / (1.f + expf(-s));
}

// ---------------------------------------------------------------------------
extern "C" void kernel_launch(void* const* d_in, const int* in_sizes, int n_in,
                              void* d_out, int out_size, void* d_ws, size_t ws_size,
                              hipStream_t stream) {
  (void)n_in; (void)out_size; (void)ws_size;
  const float* x = (const float*)d_in[0];
  int Bn = in_sizes[0] / (3 * 256 * 256);
  if (Bn <= 0) Bn = 2;

  // ---- parameter walker: JAX pytree leaf order (sorted dict keys) ----------
  int curp = 1;
  auto nf = [&]() -> const float* { return (const float*)d_in[curp++]; };
  struct Blk {
    const float *fc1_b, *fc1_w, *fc2_b, *fc2_w, *n1_b, *n1_g, *n2_b, *n2_g,
                *proj_b, *proj_w, *qkv_b, *qkv_w;
  };
  auto readBlk = [&](Blk& k) {
    k.fc1_b = nf(); k.fc1_w = nf(); k.fc2_b = nf(); k.fc2_w = nf();
    k.n1_b = nf();  k.n1_g = nf();  k.n2_b = nf();  k.n2_g = nf();
    k.proj_b = nf(); k.proj_w = nf(); k.qkv_b = nf(); k.qkv_w = nf();
  };
  Blk bot; readBlk(bot);                                    // 'bottleneck'
  Blk decBlk[4][6];
  const float *dec_fb[4] = {}, *dec_fw[4] = {}, *dec_ub[4] = {}, *dec_uw[4] = {};
  int decDepth[4] = {2, 6, 2, 2};                           // dec list: i=3,2,1,0
  for (int L = 0; L < 4; L++) {
    for (int j = 0; j < decDepth[L]; j++) readBlk(decBlk[L][j]);
    if (L > 0) { dec_fb[L] = nf(); dec_fw[L] = nf(); dec_ub[L] = nf(); dec_uw[L] = nf(); }
  }
  Blk encBlk[4][6];
  const float *enc_db[4] = {}, *enc_dw[4] = {};
  int encDepth[4] = {2, 2, 6, 2};                           // enc list: i=0..3
  for (int L = 0; L < 4; L++) {
    for (int j = 0; j < encDepth[L]; j++) readBlk(encBlk[L][j]);
    if (L > 0) { enc_db[L] = nf(); enc_dw[L] = nf(); }
  }
  const float* mask_b = nf(); const float* mask_w = nf();
  const float* pe_b   = nf(); const float* pe_w   = nf();
  const float* star_b = nf(); const float* star_w = nf();
  const float* un_b   = nf(); const float* un_w   = nf();

  // ---- workspace bump allocator --------------------------------------------
  char* base = (char*)d_ws; size_t off = 0;
  auto alloc = [&](size_t bytes) -> void* {
    off = (off + 255) & ~(size_t)255;
    void* p = base + off; off += bytes; return p;
  };
  h16* wt = (h16*)alloc(sizeof(h16) * 4u * 1024 * 1024);    // shared f16 weight buffer

  int dims[4]  = {96, 192, 384, 768};
  int Hs[4]    = {64, 32, 16, 8};
  int heads[4] = {3, 6, 12, 24};
  int Ms[4]; float* lvl[4]; float* skp[4];
  for (int i = 0; i < 4; i++) {
    Ms[i] = Bn * Hs[i] * Hs[i];
    lvl[i] = (float*)alloc(sizeof(float) * (size_t)Ms[i] * dims[i]);
    skp[i] = (float*)alloc(sizeof(float) * (size_t)Ms[i] * dims[i]);
  }
  size_t maxMC = (size_t)Ms[0] * dims[0];
  float* pat  = (float*)alloc(sizeof(float) * (size_t)Ms[0] * 64);
  float* tLn  = (float*)alloc(sizeof(float) * maxMC);
  float* tQkv = (float*)alloc(sizeof(float) * maxMC * 3);
  float* tAtt = (float*)alloc(sizeof(float) * maxMC);
  float* tH   = (float*)alloc(sizeof(float) * maxMC * 4);
  float* tCol = (float*)alloc(sizeof(float) * maxMC);
  float* tUpY = (float*)alloc(sizeof(float) * maxMC);
  float* tUp  = (float*)alloc(sizeof(float) * maxMC);
  float* tCat = (float*)alloc(sizeof(float) * maxMC * 2);
  float* tUnY = (float*)alloc(sizeof(float) * (size_t)Ms[0] * 1536);
  float* tUn  = (float*)alloc(sizeof(float) * (size_t)Bn * 256 * 256 * 96);

  // ---- launch helpers ------------------------------------------------------
  // A[M,Kpad] x W[N,Ksrc]^T ; Ksrc <= Kpad (A zero-padded); M % 64 == 0.
  auto gemm = [&](const float* A, const float* Wsrc, int N, int Ksrc, int Kpad,
                  const float* bias, int biasPeriod, const float* resid,
                  float* Cdst, int M, int act) {
    int Npad = (N + 63) & ~63;
    int tot = Kpad * Npad;
    k_wt<<<(tot + 255) / 256, 256, 0, stream>>>(Wsrc, wt, N, Ksrc, Npad, tot);
    dim3 g(Npad / 64, M / 64);
    k_gemm<<<g, 256, 0, stream>>>(A, wt, bias, resid, Cdst, M, N, Kpad, Npad,
                                  act, biasPeriod);
  };
  auto runBlock = [&](float* map, int L, const Blk& bk, int shift) {
    int M = Ms[L], C = dims[L], Hh = Hs[L];
    k_ln<<<(M + 7) / 8, 256, 0, stream>>>(map, bk.n1_g, bk.n1_b, tLn, M, C);
    gemm(tLn, bk.qkv_w, 3 * C, C, C, bk.qkv_b, 3 * C, nullptr, tQkv, M, 0);
    int nblk = Bn * (Hh / 8) * (Hh / 8) * heads[L];
    k_attn<<<nblk, 32, 0, stream>>>(tQkv, tAtt, Bn, Hh, Hh, C, heads[L], shift);
    gemm(tAtt, bk.proj_w, C, C, C, bk.proj_b, C, map, map, M, 0);
    k_ln<<<(M + 7) / 8, 256, 0, stream>>>(map, bk.n2_g, bk.n2_b, tLn, M, C);
    gemm(tLn, bk.fc1_w, 4 * C, C, C, bk.fc1_b, 4 * C, nullptr, tH, M, 1);
    gemm(tH, bk.fc2_w, C, 4 * C, 4 * C, bk.fc2_b, C, map, map, M, 0);
  };

  // ---- stem ----------------------------------------------------------------
  {
    int tot = Ms[0] * 64;
    k_patch<<<(tot + 255) / 256, 256, 0, stream>>>(x, pat, Bn, tot);
    gemm(pat, pe_w, 96, 48, 64, pe_b, 96, nullptr, lvl[0], Ms[0], 0);
  }
  // ---- encoder -------------------------------------------------------------
  for (int i = 0; i < 4; i++) {
    if (i > 0) {
      int K4 = 4 * dims[i - 1];
      int tot = Ms[i] * K4;
      k_s2d<<<(tot + 255) / 256, 256, 0, stream>>>(lvl[i - 1], tCol, Bn,
                                                   Hs[i - 1], Hs[i - 1], dims[i - 1], tot);
      gemm(tCol, enc_dw[i], dims[i], K4, K4, enc_db[i], dims[i],
           nullptr, lvl[i], Ms[i], 0);
    }
    hipMemcpyAsync(skp[i], lvl[i], sizeof(float) * (size_t)Ms[i] * dims[i],
                   hipMemcpyDeviceToDevice, stream);
    for (int j = 0; j < encDepth[i]; j++)
      runBlock(lvl[i], i, encBlk[i][j], (j & 1) ? 4 : 0);
  }
  // ---- bottleneck ----------------------------------------------------------
  runBlock(lvl[3], 3, bot, 0);
  // ---- decoder -------------------------------------------------------------
  for (int L = 0; L < 4; L++) {
    int i = 3 - L;
    if (L > 0) {
      int Cin = dims[i + 1], O = dims[i];
      int Npad = 4 * O;                         // 384/768/1536: already %64
      int totw = Cin * Npad;
      k_wt_up<<<(totw + 255) / 256, 256, 0, stream>>>(dec_uw[L], wt, Cin, O, 4,
                                                      Npad, totw);
      dim3 g(Npad / 64, Ms[i + 1] / 64);
      k_gemm<<<g, 256, 0, stream>>>(lvl[i + 1], wt, dec_ub[L], nullptr, tUpY,
                                    Ms[i + 1], 4 * O, Cin, Npad, 0, O);
      int tot = Ms[i] * O;
      k_upsc<<<(tot + 255) / 256, 256, 0, stream>>>(tUpY, tUp, Bn, Hs[i + 1], Hs[i + 1],
                                                    O, 2, tot);
      int totc = Ms[i] * 2 * O;
      k_cat<<<(totc + 255) / 256, 256, 0, stream>>>(tUp, skp[i], tCat, Ms[i], O, O, totc);
      gemm(tCat, dec_fw[L], O, 2 * O, 2 * O, dec_fb[L], O, nullptr, lvl[i], Ms[i], 0);
    }
    for (int j = 0; j < decDepth[L]; j++)
      runBlock(lvl[i], i, decBlk[L][j], (j & 1) ? 4 : 0);
  }
  // ---- un-patchify (ConvTranspose 4x4 s4 as GEMM + scatter) ----------------
  {
    int totw = 96 * 1536;
    k_wt_up<<<(totw + 255) / 256, 256, 0, stream>>>(un_w, wt, 96, 96, 16, 1536, totw);
    dim3 g(1536 / 64, Ms[0] / 64);
    k_gemm<<<g, 256, 0, stream>>>(lvl[0], wt, un_b, nullptr, tUnY, Ms[0], 1536, 96,
                                  1536, 0, 96);
    int tot = Bn * 256 * 256 * 96;
    k_upsc<<<(tot + 255) / 256, 256, 0, stream>>>(tUnY, tUn, Bn, 64, 64, 96, 4, tot);
  }
  // ---- heads ---------------------------------------------------------------
  float* outp = (float*)d_out;
  {
    int tot = Bn * 3 * 256 * 256;
    k_conv3<<<(tot + 255) / 256, 256, 0, stream>>>(tUn, star_w, star_b, outp,
                                                   Bn, 256, 256, 96, 3, tot);
    int tot2 = Bn * 1 * 256 * 256;
    k_conv3<<<(tot2 + 255) / 256, 256, 0, stream>>>(tUn, mask_w, mask_b, outp + tot,
                                                    Bn, 256, 256, 96, 1, tot2);
  }
}
